// Attention_26989574488169
// MI455X (gfx1250) — matrix operand order repaired and checked by reading, not compiled, not run on hardware
//
#include <hip/hip_runtime.h>
#include <hip/hip_bf16.h>

#define NN   50000
#define PP   4
#define EE   800000
#define INS  256
#define OUTS 128
#define WSTRIDE 144   // padded K-stride (halfwords) for transposed W tile in LDS

typedef __attribute__((ext_vector_type(16))) __bf16 v16bf;
typedef __attribute__((ext_vector_type(8)))  float  v8f;

#define CDIV(a,b) (((a)+(b)-1)/(b))

// Identity K order within each 32-wide group (contiguous k = 16*laneHalf + s),
// matching the contiguous B fragments; each lane's fragment is one 32-byte chunk.
__device__ __forceinline__ int aperm(int r) {
    // r in [0,32)
    return r;
}

// ---------------- utility kernels ----------------

__global__ void zero_f(float* p, int n) {
    int i = blockIdx.x * blockDim.x + threadIdx.x;
    if (i < n) p[i] = 0.0f;
}

__global__ void degrees_kernel(const int* __restrict__ src, const int* __restrict__ dst,
                               float* __restrict__ deg_out, float* __restrict__ deg_in) {
    int i = blockIdx.x * blockDim.x + threadIdx.x;
    if (i >= PP * EE) return;
    int p = i / EE;
    atomicAdd(&deg_out[p * NN + src[i]], 1.0f);
    atomicAdd(&deg_in [p * NN + dst[i]], 1.0f);
}

__global__ void rsqrt_kernel(float* d, int n) {
    int i = blockIdx.x * blockDim.x + threadIdx.x;
    if (i < n) d[i] = rsqrtf(fmaxf(d[i], 1.0f));
}

// x (f32, row-major [N][256]) -> xp (bf16, A-fragment-permuted within 32-K groups)
__global__ void cvt_perm_kernel(const float* __restrict__ x, __bf16* __restrict__ xp) {
    int i = blockIdx.x * blockDim.x + threadIdx.x;
    if (i >= NN * INS) return;
    const int k = i & (INS - 1);
    const int n = i >> 8;               // INS = 256
    const int g = k >> 5;
    const int r = k & 31;
    xp[(size_t)n * INS + g * 32 + aperm(r)] = (__bf16)x[i];
}

// ---------------- GEMM 1: y = x(bf16,permuted) @ W[p]  (50000x256 @ 256x128) ----
// 256 threads = 8 waves; block computes 128 rows x 128 cols; wave = 16-row strip.
__global__ __launch_bounds__(256)
void gemm_xw_kernel(const __bf16* __restrict__ xp, const float* __restrict__ Wg,
                    float* __restrict__ y, int p) {
    // transposed W tile: Wt[n*WSTRIDE + k], one 128-deep K half at a time (36 KB)
    __shared__ __align__(32) __bf16 Wt[OUTS * WSTRIDE];
    const float* Wp = Wg + (size_t)p * INS * OUTS;

    const int wave = threadIdx.x >> 5;
    const int lane = threadIdx.x & 31;
    const int laneLo = lane & 15;
    const int laneHalf = lane >> 4;
    const int m0 = blockIdx.x * 128 + wave * 16;

    int arow = m0 + laneLo;
    if (arow >= NN) arow = NN - 1;                    // clamp; masked on store
    const __bf16* arowp = xp + (size_t)arow * INS;

    v8f c[8];
#pragma unroll
    for (int t = 0; t < 8; ++t)
#pragma unroll
        for (int r = 0; r < 8; ++r) c[t][r] = 0.0f;

#pragma unroll
    for (int kh = 0; kh < 2; ++kh) {                  // two K halves of 128
        // stage W[k][n] -> Wt[n][k] (bf16)
        for (int i = threadIdx.x; i < 128 * OUTS; i += 256) {
            const int k = i >> 7;
            const int n = i & 127;
            Wt[n * WSTRIDE + k] = (__bf16)Wp[kh * 128 * OUTS + i];
        }
        __syncthreads();

#pragma unroll
        for (int kt = 0; kt < 4; ++kt) {
            // A fragment: contiguous 32B thanks to permuted layout
            v16bf a = *(const v16bf*)(arowp + kh * 128 + kt * 32 + laneHalf * 16);
            const int kB = kt * 32 + laneHalf * 16;   // local K base for B
#pragma unroll
            for (int nt = 0; nt < 8; ++nt) {
                const int col = nt * 16 + laneLo;
                v16bf b = *(const v16bf*)(&Wt[col * WSTRIDE + kB]);
                c[nt] = __builtin_amdgcn_wmma_f32_16x16x32_bf16(
                    false, a, false, b, (short)0, c[nt], false, false);
            }
        }
        __syncthreads();
    }

#pragma unroll
    for (int nt = 0; nt < 8; ++nt) {
        const int n = nt * 16 + laneLo;
#pragma unroll
        for (int r = 0; r < 8; ++r) {
            const int m = m0 + laneHalf * 8 + r;
            if (m < NN) y[(size_t)m * OUTS + n] = c[nt][r];
        }
    }
}

// ---------------- edge aggregation: agg[p][dst] += y[src] * rs_out[p][src] -----
// one wave per edge; lane owns 4 consecutive columns (float4 gather, f32 atomics)
__global__ __launch_bounds__(256)
void aggregate_kernel(const float* __restrict__ y, const int* __restrict__ src,
                      const int* __restrict__ dst, const float* __restrict__ rs_out,
                      float* __restrict__ agg, int p) {
    const int wave = threadIdx.x >> 5;
    const int lane = threadIdx.x & 31;
    const int e = blockIdx.x * 8 + wave;
    if (e >= EE) return;
    const int s = src[(size_t)p * EE + e];
    const int d = dst[(size_t)p * EE + e];
    const float scale = rs_out[p * NN + s];
    const float4 v = *(const float4*)(y + (size_t)s * OUTS + lane * 4);
    float* ad = agg + ((size_t)p * NN + d) * OUTS + lane * 4;
    atomicAdd(&ad[0], v.x * scale);
    atomicAdd(&ad[1], v.y * scale);
    atomicAdd(&ad[2], v.z * scale);
    atomicAdd(&ad[3], v.w * scale);
}

// ---------------- h = agg * rs_in + b_conv (in place) + bf16 permuted copy -----
__global__ void finish_h_kernel(float* __restrict__ h, const float* __restrict__ rs_in,
                                const float* __restrict__ b_conv, __bf16* __restrict__ hb) {
    int idx = blockIdx.x * blockDim.x + threadIdx.x;
    if (idx >= PP * NN * OUTS) return;
    const int c = idx & (OUTS - 1);
    const int pn = idx >> 7;            // OUTS = 128
    const int n = pn % NN;
    const int p = pn / NN;
    const float v = h[idx] * rs_in[p * NN + n] + b_conv[p * OUTS + c];
    h[idx] = v;
    const int g = c >> 5;
    const int r = c & 31;
    hb[(size_t)pn * OUTS + g * 32 + aperm(r)] = (__bf16)v;
}

// ---------------- GEMM 2 + tanh + column-sum partials --------------------------
// z = h_p @ fc_w^T ; sp[p][j] += sum_rows tanh(z + fc_b[j])
__global__ __launch_bounds__(256)
void gemm_att_kernel(const __bf16* __restrict__ hb, const float* __restrict__ fcw,
                     const float* __restrict__ fcb, float* __restrict__ sp, int p) {
    __shared__ __align__(32) __bf16 Fl[OUTS * OUTS];  // Fl[j*128+k] = fc_w[j][k]
    for (int i = threadIdx.x; i < OUTS * OUTS; i += 256) Fl[i] = (__bf16)fcw[i];
    __syncthreads();

    const int wave = threadIdx.x >> 5;
    const int lane = threadIdx.x & 31;
    const int laneLo = lane & 15;
    const int laneHalf = lane >> 4;
    const int m0 = blockIdx.x * 128 + wave * 16;

    int arow = m0 + laneLo;
    if (arow >= NN) arow = NN - 1;
    const __bf16* arowp = hb + ((size_t)p * NN + arow) * OUTS;

    v8f c[8];
#pragma unroll
    for (int t = 0; t < 8; ++t)
#pragma unroll
        for (int r = 0; r < 8; ++r) c[t][r] = 0.0f;

#pragma unroll
    for (int kt = 0; kt < 4; ++kt) {                  // K = 128
        v16bf a = *(const v16bf*)(arowp + kt * 32 + laneHalf * 16);
        const int kB = kt * 32 + laneHalf * 16;
#pragma unroll
        for (int nt = 0; nt < 8; ++nt) {
            const int n = nt * 16 + laneLo;
            // B[k][n] = fc_w[n][k] -> contiguous 16 bf16 in LDS
            v16bf b = *(const v16bf*)(&Fl[n * OUTS + kB]);
            c[nt] = __builtin_amdgcn_wmma_f32_16x16x32_bf16(
                false, a, false, b, (short)0, c[nt], false, false);
        }
    }

#pragma unroll
    for (int nt = 0; nt < 8; ++nt) {
        const int n = nt * 16 + laneLo;
        const float bias = fcb[n];
        float ssum = 0.0f;
#pragma unroll
        for (int r = 0; r < 8; ++r) {
            const int m = m0 + laneHalf * 8 + r;
            if (m < NN) ssum += tanhf(c[nt][r] + bias);
        }
        atomicAdd(&sp[p * OUTS + n], ssum);
    }
}

// ---------------- beta = softmax_p( att[p] . sp[p]/N ) -------------------------
__global__ void beta_kernel(const float* __restrict__ att, const float* __restrict__ sp,
                            float* __restrict__ beta) {
    __shared__ float red[128];
    __shared__ float sv[PP];
    const int j = threadIdx.x;
    for (int p = 0; p < PP; ++p) {
        red[j] = att[p * OUTS + j] * sp[p * OUTS + j] * (1.0f / (float)NN);
        __syncthreads();
        for (int s = 64; s > 0; s >>= 1) {
            if (j < s) red[j] += red[j + s];
            __syncthreads();
        }
        if (j == 0) sv[p] = red[0];
        __syncthreads();
    }
    if (j == 0) {
        float mx = sv[0];
        for (int p = 1; p < PP; ++p) mx = fmaxf(mx, sv[p]);
        float e[PP], sum = 0.0f;
        for (int p = 0; p < PP; ++p) { e[p] = __expf(sv[p] - mx); sum += e[p]; }
        for (int p = 0; p < PP; ++p) beta[p] = e[p] / sum;
    }
}

// ---------------- out = sum_p beta[p]*h[p] + h_bias ----------------------------
__global__ void combine_kernel(const float* __restrict__ h, const float* __restrict__ beta,
                               const float* __restrict__ h_bias, float* __restrict__ out) {
    int idx = blockIdx.x * blockDim.x + threadIdx.x;
    if (idx >= NN * OUTS) return;
    const int c = idx & (OUTS - 1);
    const size_t st = (size_t)NN * OUTS;
    float v = h[idx] * beta[0] + h[idx + st] * beta[1]
            + h[idx + 2 * st] * beta[2] + h[idx + 3 * st] * beta[3];
    out[idx] = v + h_bias[c];
}

// ---------------- launch -------------------------------------------------------
extern "C" void kernel_launch(void* const* d_in, const int* in_sizes, int n_in,
                              void* d_out, int out_size, void* d_ws, size_t ws_size,
                              hipStream_t stream) {
    const float* x      = (const float*)d_in[0];   // [N, IN]
    const float* W      = (const float*)d_in[1];   // [P, IN, OUT]
    const float* b_conv = (const float*)d_in[2];   // [P, OUT]
    const float* att    = (const float*)d_in[3];   // [P, OUT]
    const float* fc_w   = (const float*)d_in[4];   // [OUT, OUT]
    const float* fc_b   = (const float*)d_in[5];   // [OUT]
    const float* h_bias = (const float*)d_in[6];   // [OUT]
    const int*   src    = (const int*)d_in[7];     // [P, E]
    const int*   dst    = (const int*)d_in[8];     // [P, E]
    float* out = (float*)d_out;

    // workspace layout (float units)
    float*  ws      = (float*)d_ws;
    float*  deg_out = ws;                               // P*N   (becomes rs_out)
    float*  deg_in  = deg_out + PP * NN;                // P*N   (becomes rs_in)
    __bf16* xp      = (__bf16*)(deg_in + PP * NN);      // N*IN bf16 (permuted)
    float*  y       = (float*)(xp + (size_t)NN * INS);  // N*OUT (per-path, reused)
    float*  agg     = y + (size_t)NN * OUTS;            // P*N*OUT (becomes h)
    float*  sp      = agg + (size_t)PP * NN * OUTS;     // P*OUT
    float*  beta    = sp + PP * OUTS;                   // P
    // hb overlays xp..y (both dead once aggregation is done):
    // needs P*N*OUT bf16 = 51.2 MB == size of (xp 25.6 MB + y 25.6 MB)
    __bf16* hb      = xp;

    // 1) zero accumulators
    zero_f<<<CDIV(2 * PP * NN, 256), 256, 0, stream>>>(deg_out, 2 * PP * NN);
    zero_f<<<CDIV(PP * NN * OUTS + PP * OUTS + PP, 256), 256, 0, stream>>>(
        agg, PP * NN * OUTS + PP * OUTS + PP);

    // 2) degrees + rsqrt (in place)
    degrees_kernel<<<CDIV(PP * EE, 256), 256, 0, stream>>>(src, dst, deg_out, deg_in);
    rsqrt_kernel<<<CDIV(2 * PP * NN, 256), 256, 0, stream>>>(deg_out, 2 * PP * NN);

    // 3) x -> bf16, A-fragment-permuted (deg_out scale folded into edge gather)
    cvt_perm_kernel<<<CDIV(NN * INS, 256), 256, 0, stream>>>(x, xp);

    // 4) per meta-path: WMMA GEMM then L2-resident edge scatter-add
    const int gemm_grid = CDIV(NN, 128);
    for (int p = 0; p < PP; ++p) {
        gemm_xw_kernel<<<gemm_grid, 256, 0, stream>>>(xp, W, y, p);
        aggregate_kernel<<<CDIV(EE, 8), 256, 0, stream>>>(y, src, dst, deg_out, agg, p);
    }

    // 5) h = agg * rsqrt(deg_in) + b_conv  (+ bf16 permuted copy for GEMM2)
    finish_h_kernel<<<CDIV(PP * NN * OUTS, 256), 256, 0, stream>>>(agg, deg_in, b_conv, hb);

    // 6) semantic attention: tanh(h @ fc_w^T + fc_b) column sums
    for (int p = 0; p < PP; ++p)
        gemm_att_kernel<<<gemm_grid, 256, 0, stream>>>(hb, fc_w, fc_b, sp, p);

    // 7) beta softmax + final combine
    beta_kernel<<<1, 128, 0, stream>>>(att, sp, beta);
    combine_kernel<<<CDIV(NN * OUTS, 256), 256, 0, stream>>>(agg, beta, h_bias, out);
}
